// MambaClassifier_49632642072978
// MI455X (gfx1250) — compile-verified, matrix operand order checked by reading, not run on hardware
//
#include <hip/hip_runtime.h>

#define EPSF 1e-5f

typedef __attribute__((ext_vector_type(16))) _Float16 v16h;
typedef __attribute__((ext_vector_type(8)))  float    v8f;

union FragAB { v16h v; unsigned int u[8]; };

__device__ inline unsigned int pack_f16x2(float a, float b) {
  union { _Float16 h[2]; unsigned int u; } cv;
  cv.h[0] = (_Float16)a; cv.h[1] = (_Float16)b;
  return cv.u;
}

// ---------------------------------------------------------------------------
// Generalized batched GEMM on the WMMA pipe.
//   wT==0 (TN): C[m,n] = scale * sum_k A[m*lda+k] * W[n*ldw+k]   (+bias)
//   wT==1 (NN): C[m,n] = scale * sum_k A[m*lda+k] * W[k*ldw+n]   (+bias)
// Batch: z = blockIdx.z, z1 = z>>zShift, z0 = z & ((1<<zShift)-1);
//   A += z1*aZ1 + z0*aZ0   (same for W with wZ*, C with cZ*)
// Requirements: K%32==0, M%64==0. N-edge handled branchlessly.
// Block tile 64M x 128N, 8 waves, 4 v_wmma_f32_16x16x32_f16 per K-step,
// next K-tile's global loads issued under the current step's WMMAs.
// ---------------------------------------------------------------------------
__global__ __launch_bounds__(256) void gemm_wmma_ex(
    const float* __restrict__ A, const float* __restrict__ W,
    const float* __restrict__ bias, float* __restrict__ C,
    int M, int N, int K, int lda, int ldw, int ldc,
    int zShift, long aZ1, long aZ0, long wZ1, long wZ0, long cZ1, long cZ0,
    float scale, int act, int wT)
{
  __shared__ unsigned int sA[64 * 16];   // 64 rows x 32 f16 (16 u32/row)
  __shared__ unsigned int sB[128 * 16];  // 128 cols x 32 f16
  int tid  = threadIdx.x;
  int wave = tid >> 5;
  int lane = tid & 31;
  int m0    = blockIdx.y * 64;
  int n0blk = blockIdx.x * 128;
  int half  = lane >> 4;
  int r15   = lane & 15;

  long z  = blockIdx.z;
  long z1 = z >> zShift;
  long z0 = z & ((1 << zShift) - 1);
  const float* Ab = A + z1 * aZ1 + z0 * aZ0;
  const float* Wb = W + z1 * wZ1 + z0 * wZ0;
  float*       Cb = C + z1 * cZ1 + z0 * cZ0;

  // Staging assignments: A = 1024 u32 entries (4/thread), B = 2048 (8/thread)
  long aOff[4];
  #pragma unroll
  for (int it = 0; it < 4; ++it) {
    int idx = tid + it * 256;
    aOff[it] = (long)(m0 + (idx >> 4)) * lda + 2 * (idx & 15);
  }
  long bOff[8];
  float bMask[8];
  int sBi[8];
  if (!wT) {
    #pragma unroll
    for (int it = 0; it < 8; ++it) {
      int idx = tid + it * 256;
      int gn = n0blk + (idx >> 4);
      bMask[it] = (gn < N) ? 1.f : 0.f;
      if (gn > N - 1) gn = N - 1;
      bOff[it] = (long)gn * ldw + 2 * (idx & 15);
      sBi[it] = idx;
    }
  } else {
    #pragma unroll
    for (int it = 0; it < 8; ++it) {
      int idx = tid + it * 256;
      int n = idx & 127, kp = idx >> 7;
      int gn = n0blk + n;
      bMask[it] = (gn < N) ? 1.f : 0.f;
      if (gn > N - 1) gn = N - 1;
      bOff[it] = (long)(2 * kp) * ldw + gn;
      sBi[it] = n * 16 + kp;
    }
  }

  v8f acc[4];
  #pragma unroll
  for (int mt = 0; mt < 4; ++mt) acc[mt] = (v8f){};

  // prefetch first K-tile
  float2 aReg[4], bReg[8];
  #pragma unroll
  for (int it = 0; it < 4; ++it) aReg[it] = *(const float2*)(Ab + aOff[it]);
  if (!wT) {
    #pragma unroll
    for (int it = 0; it < 8; ++it) bReg[it] = *(const float2*)(Wb + bOff[it]);
  } else {
    #pragma unroll
    for (int it = 0; it < 8; ++it) {
      bReg[it].x = Wb[bOff[it]];
      bReg[it].y = Wb[bOff[it] + ldw];
    }
  }

  for (int k0 = 0; k0 < K; k0 += 32) {
    // commit staged registers to LDS
    #pragma unroll
    for (int it = 0; it < 4; ++it)
      sA[tid + it * 256] = pack_f16x2(aReg[it].x, aReg[it].y);
    #pragma unroll
    for (int it = 0; it < 8; ++it)
      sB[sBi[it]] = pack_f16x2(bReg[it].x * bMask[it], bReg[it].y * bMask[it]);
    __syncthreads();

    // issue next K-tile's loads (hidden under the WMMAs below)
    int kn = k0 + 32;
    if (kn < K) {
      #pragma unroll
      for (int it = 0; it < 4; ++it)
        aReg[it] = *(const float2*)(Ab + aOff[it] + kn);
      if (!wT) {
        #pragma unroll
        for (int it = 0; it < 8; ++it)
          bReg[it] = *(const float2*)(Wb + bOff[it] + kn);
      } else {
        #pragma unroll
        for (int it = 0; it < 8; ++it) {
          bReg[it].x = Wb[bOff[it] + (long)kn * ldw];
          bReg[it].y = Wb[bOff[it] + (long)kn * ldw + ldw];
        }
      }
    }

    // B fragment reused by 4 M-subtiles
    FragAB fb;
    #pragma unroll
    for (int i = 0; i < 8; ++i) {
      int k2 = ((i < 4) ? 0 : 8) + (i & 3) + half * 4;
      fb.u[i] = sB[(wave * 16 + r15) * 16 + k2];
    }
    #pragma unroll
    for (int mt = 0; mt < 4; ++mt) {
      FragAB fa;
      #pragma unroll
      for (int i = 0; i < 8; ++i) {
        int k2 = ((i < 4) ? 0 : 8) + (i & 3) + half * 4;
        fa.u[i] = sA[(mt * 16 + r15) * 16 + k2];
      }
      acc[mt] = __builtin_amdgcn_wmma_f32_16x16x32_f16(
          false, fa.v, false, fb.v, (short)0, acc[mt], false, false);
    }
    __syncthreads();
  }

  int gn = n0blk + wave * 16 + r15;
  if (gn < N) {
    float bv = bias ? bias[gn] : 0.f;
    #pragma unroll
    for (int mt = 0; mt < 4; ++mt) {
      #pragma unroll
      for (int r = 0; r < 8; ++r) {
        int gm = m0 + mt * 16 + r + half * 8;
        if (gm < M) {
          float v = acc[mt][r] * scale + bv;
          if (act == 1) v = fmaxf(v, 0.f);
          Cb[(size_t)gm * ldc + gn] = v;
        }
      }
    }
  }
}

// ---------------------------------------------------------------------------
// Conv1d k=3 pad=1 on (B,Cin,L) -> (B,Cout,L), optional fused eval batchnorm
// ---------------------------------------------------------------------------
__global__ void conv1d_bn_kernel(
    const float* __restrict__ x, const float* __restrict__ w,
    const float* __restrict__ bias,
    const float* g, const float* bb, const float* bm, const float* bv,
    float* __restrict__ y, int B, int Cin, int Cout, int L)
{
  int idx = blockIdx.x * 256 + threadIdx.x;
  if (idx >= B * Cout * L) return;
  int l = idx % L;
  int o = (idx / L) % Cout;
  int b = idx / (L * Cout);
  float acc = bias[o];
  const float* xb = x + (size_t)b * Cin * L;
  const float* wo = w + (size_t)o * Cin * 3;
  for (int i = 0; i < Cin; ++i) {
    float x0 = (l >= 1)     ? xb[i * L + l - 1] : 0.f;
    float x1 =                xb[i * L + l];
    float x2 = (l + 1 < L)  ? xb[i * L + l + 1] : 0.f;
    acc += wo[i * 3 + 0] * x0 + wo[i * 3 + 1] * x1 + wo[i * 3 + 2] * x2;
  }
  if (g) acc = (acc - bm[o]) * rsqrtf(bv[o] + EPSF) * g[o] + bb[o];
  y[idx] = acc;
}

__global__ void elu_add_kernel(const float* a, const float* b, float* out, int n) {
  int idx = blockIdx.x * 256 + threadIdx.x;
  if (idx >= n) return;
  float v = a[idx] + b[idx];
  out[idx] = (v > 0.f) ? v : expm1f(v);
}

// (B,L,3) -> (B,3,L)
__global__ void in_transpose_kernel(const float* x, float* xt) {
  int idx = blockIdx.x * 256 + threadIdx.x;
  if (idx >= 2 * 3 * 1024) return;
  int l = idx & 1023;
  int c = (idx >> 10) % 3;
  int b = idx / 3072;
  xt[idx] = x[((size_t)b * 1024 + l) * 3 + c];
}

// (B,256,L) -> tokens (B*L,256)
__global__ void to_tokens_kernel(const float* x, float* h) {
  int idx = blockIdx.x * 256 + threadIdx.x; // (b*1024+l)*256+c
  int c = idx & 255;
  int l = (idx >> 8) & 1023;
  int b = idx >> 18;
  h[idx] = x[((size_t)b * 256 + c) * 1024 + l];
}

// dt = softplus(raw + dt_bias)
__global__ void dt_kernel(const float* __restrict__ zx,
                          const float* __restrict__ dt_bias,
                          float* __restrict__ dtp) {
  int idx = blockIdx.x * 256 + threadIdx.x; // t*64+h
  int h = idx & 63;
  long t = idx >> 6;
  float x = zx[t * 8512 + 8448 + h] + dt_bias[h];
  dtp[idx] = (x > 20.f) ? x : log1pf(expf(x));
}

// causal depthwise conv (kernel 4) over xBC columns + silu
__global__ void mamba_conv_kernel(const float* __restrict__ zx,
                                  const float* __restrict__ cw,
                                  const float* __restrict__ cb,
                                  float* __restrict__ xbc) {
  long idx = (long)blockIdx.x * 256 + threadIdx.x;
  if (idx >= (long)2048 * 4352) return;
  int c = (int)(idx % 4352);
  long t = idx / 4352;
  int l = (int)(t & 1023);
  float acc = cb[c];
  #pragma unroll
  for (int j = 0; j < 4; ++j) {
    int ll = l - 3 + j;
    if (ll >= 0) acc += cw[c * 4 + j] * zx[(t - 3 + j) * 8512 + 4096 + c];
  }
  xbc[idx] = acc / (1.f + expf(-acc));
}

// Chunked SSD per (b,h): sequential over 16 chunks; state 64x128 in LDS.
__global__ __launch_bounds__(256) void ssd_kernel(
    const float* __restrict__ xbc, const float* __restrict__ dtp,
    const float* __restrict__ gbuf, const float* __restrict__ A_log,
    const float* __restrict__ Dp, float* __restrict__ y)
{
  int h = blockIdx.x & 63;
  int b = blockIdx.x >> 6;
  int tid = threadIdx.x;
  __shared__ float sS[64 * 128];
  __shared__ float sGL[64 * 64];
  __shared__ float sAc[64];
  __shared__ float sDt[64];
  __shared__ float sDd[64];
  float Ah = -expf(A_log[h]);
  float Dh = Dp[h];
  for (int i = tid; i < 64 * 128; i += 256) sS[i] = 0.f;
  for (int ch = 0; ch < 16; ++ch) {
    long t0 = (long)b * 1024 + (long)ch * 64;
    __syncthreads();
    if (tid < 64) sDt[tid] = dtp[(t0 + tid) * 64 + h];
    __syncthreads();
    if (tid == 0) {
      float c = 0.f;
      for (int j = 0; j < 64; ++j) { c += sDt[j] * Ah; sAc[j] = c; }
    }
    __syncthreads();
    float alast = sAc[63];
    if (tid < 64) sDd[tid] = expf(alast - sAc[tid]) * sDt[tid];
    const float* G = gbuf + ((long)(b * 16 + ch)) * 4096;
    for (int e = tid; e < 4096; e += 256) {
      int i = e >> 6, j = e & 63;
      sGL[e] = (j <= i) ? G[e] * expf(sAc[i] - sAc[j]) * sDt[j] : 0.f;
    }
    __syncthreads();
    // Y = diag + off-chunk + D*x
    for (int e = tid; e < 4096; e += 256) {
      int i = e >> 6, p = e & 63;
      float acc = 0.f;
      for (int j = 0; j <= i; ++j)
        acc += sGL[i * 64 + j] * xbc[(t0 + j) * 4352 + h * 64 + p];
      float off = 0.f;
      const float* Ci = xbc + (t0 + i) * 4352 + 4224;
      for (int n = 0; n < 128; ++n) off += Ci[n] * sS[p * 128 + n];
      float xi = xbc[(t0 + i) * 4352 + h * 64 + p];
      acc += expf(sAc[i]) * off + Dh * xi;
      y[(t0 + i) * 4096 + h * 64 + p] = acc;
    }
    __syncthreads();
    // S = exp(alast)*S + sum_j B[j]^T (decay*dt*x)[j]
    float cs = expf(alast);
    for (int e = tid; e < 64 * 128; e += 256) {
      int p = e >> 7, n = e & 127;
      float acc = cs * sS[e];
      for (int j = 0; j < 64; ++j)
        acc += xbc[(t0 + j) * 4352 + 4096 + n] * sDd[j] *
               xbc[(t0 + j) * 4352 + h * 64 + p];
      sS[e] = acc;
    }
  }
}

// y = (y * silu(z)) normalized by rms over 4096, scaled by norm_w
__global__ __launch_bounds__(256) void gate_rms_kernel(
    float* __restrict__ y, const float* __restrict__ zx,
    const float* __restrict__ nw)
{
  long t = blockIdx.x;
  int tid = threadIdx.x;
  __shared__ float red[256];
  float vals[16];
  float sum = 0.f;
  #pragma unroll
  for (int r = 0; r < 16; ++r) {
    int d = tid + r * 256;
    float z = zx[t * 8512 + d];
    float v = y[t * 4096 + d] * (z / (1.f + expf(-z)));
    vals[r] = v;
    sum += v * v;
  }
  red[tid] = sum; __syncthreads();
  for (int s = 128; s > 0; s >>= 1) { if (tid < s) red[tid] += red[tid + s]; __syncthreads(); }
  float scale = rsqrtf(red[0] / 4096.f + EPSF);
  #pragma unroll
  for (int r = 0; r < 16; ++r) {
    int d = tid + r * 256;
    y[t * 4096 + d] = vals[r] * scale * nw[d];
  }
}

// eval batchnorm over channel dim of (T,256)
__global__ void bn_lc_kernel(const float* __restrict__ x, const float* g,
                             const float* bb, const float* bm, const float* bv,
                             float* __restrict__ out) {
  int idx = blockIdx.x * 256 + threadIdx.x;
  int c = idx & 255;
  out[idx] = (x[idx] - bm[c]) * rsqrtf(bv[c] + EPSF) * g[c] + bb[c];
}

// softmax over rows of 1024 (in place)
__global__ __launch_bounds__(256) void softmax_row_kernel(float* __restrict__ s) {
  long row = blockIdx.x;
  float* p = s + row * 1024;
  int tid = threadIdx.x;
  __shared__ float red[256];
  float v[4];
  float lmax = -1e30f;
  #pragma unroll
  for (int r = 0; r < 4; ++r) { v[r] = p[tid + r * 256]; lmax = fmaxf(lmax, v[r]); }
  red[tid] = lmax; __syncthreads();
  for (int st = 128; st > 0; st >>= 1) { if (tid < st) red[tid] = fmaxf(red[tid], red[tid + st]); __syncthreads(); }
  float mx = red[0];
  __syncthreads();
  float sum = 0.f;
  #pragma unroll
  for (int r = 0; r < 4; ++r) { v[r] = expf(v[r] - mx); sum += v[r]; }
  red[tid] = sum; __syncthreads();
  for (int st = 128; st > 0; st >>= 1) { if (tid < st) red[tid] += red[tid + st]; __syncthreads(); }
  float inv = 1.f / red[0];
  #pragma unroll
  for (int r = 0; r < 4; ++r) p[tid + r * 256] = v[r] * inv;
}

// out = LN(a + b) * g + be   (per 256-dim token)
__global__ __launch_bounds__(256) void add_ln_kernel(
    const float* __restrict__ a, const float* __restrict__ b,
    const float* g, const float* be, float* __restrict__ out)
{
  long t = blockIdx.x;
  int tid = threadIdx.x;
  __shared__ float red[256];
  float v = a[t * 256 + tid] + b[t * 256 + tid];
  red[tid] = v; __syncthreads();
  for (int s = 128; s > 0; s >>= 1) { if (tid < s) red[tid] += red[tid + s]; __syncthreads(); }
  float mu = red[0] / 256.f;
  __syncthreads();
  float d = v - mu;
  red[tid] = d * d; __syncthreads();
  for (int s = 128; s > 0; s >>= 1) { if (tid < s) red[tid] += red[tid + s]; __syncthreads(); }
  float var = red[0] / 256.f;
  out[t * 256 + tid] = d * rsqrtf(var + EPSF) * g[tid] + be[tid];
}

// final 128 -> 9 projection
__global__ void outproj_kernel(const float* __restrict__ fc,
                               const float* __restrict__ w,
                               const float* __restrict__ bias,
                               float* __restrict__ out) {
  int idx = blockIdx.x * 256 + threadIdx.x;
  if (idx >= 2048 * 9) return;
  int j = idx % 9;
  long t = idx / 9;
  float acc = bias[j];
  for (int k = 0; k < 128; ++k) acc += fc[t * 128 + k] * w[j * 128 + k];
  out[idx] = acc;
}

// ---------------------------------------------------------------------------
extern "C" void kernel_launch(void* const* d_in, const int* in_sizes, int n_in,
                              void* d_out, int out_size, void* d_ws, size_t ws_size,
                              hipStream_t stream)
{
  const float* const* in = (const float* const*)d_in;
  const float* X = in[0];
  const float* convd_w = in[1];  const float* convd_b = in[2];
  const float* conv1d_w = in[3]; const float* conv1d_b = in[4];
  const float* c1_w = in[5];  const float* c1_b = in[6];
  const float* c2_w = in[11]; const float* c2_b = in[12];
  const float* c3_w = in[17]; const float* c3_b = in[18];
  const float* c4_w = in[23]; const float* c4_b = in[24];
  // bnc1: 7..10, bnc2: 13..16, bnc3: 19..22, bnc4: 25..28
  // mamba layers: base 29 + 8*i : {in_w, conv_w, conv_b, dt_bias, A_log, D, norm_w, out_w}
  // bn1..bn4: base 61 + 4*i : {g,b,m,v}
  const float* attn_in_w  = in[77]; const float* attn_in_b  = in[78];
  const float* attn_out_w = in[79]; const float* attn_out_b = in[80];
  const float* ln_g = in[81]; const float* ln_b = in[82];
  const float* fc_w = in[83]; const float* fc_b = in[84];
  const float* ow   = in[85]; const float* ob   = in[86];

  float* ws = (float*)d_ws;
  size_t off = 0;
  auto alloc = [&](size_t n) { float* p = ws + off; off += (n + 63) & ~(size_t)63; return p; };
  float* xT     = alloc((size_t)2 * 3 * 1024);
  float* c64    = alloc((size_t)2 * 64 * 1024);
  float* c128   = alloc((size_t)2 * 128 * 1024);
  float* cb1    = alloc((size_t)2 * 256 * 1024);
  float* cb2    = alloc((size_t)2 * 256 * 1024);
  float* htok   = alloc((size_t)2048 * 256);
  float* zx     = alloc((size_t)2048 * 8512);
  float* xbc    = alloc((size_t)2048 * 4352);
  float* dtp    = alloc((size_t)2048 * 64);
  float* gbuf   = alloc((size_t)2 * 16 * 64 * 64);
  float* ybig   = alloc((size_t)2048 * 4096);
  float* tmp256 = alloc((size_t)2048 * 256);
  float* fcbuf  = alloc((size_t)2048 * 128);
  float* sbuf   = alloc((size_t)2 * 8 * 1024 * 1024);  // attention scores
  (void)ws_size; (void)in_sizes; (void)n_in;

  const float* NUL = nullptr;

  // plain (unbatched) TN GEMM
  auto gemm = [&](const float* A, const float* W, const float* bias, float* C,
                  int M, int N, int K, int act) {
    dim3 grid((N + 127) / 128, M / 64, 1);
    gemm_wmma_ex<<<grid, 256, 0, stream>>>(A, W, bias, C, M, N, K, K, K, N,
                                           0, 0, 0, 0, 0, 0, 0, 1.0f, act, 0);
  };

  // ---- conv front-end ----
  in_transpose_kernel<<<(2 * 3 * 1024 + 255) / 256, 256, 0, stream>>>(X, xT);
  conv1d_bn_kernel<<<(2 * 64 * 1024 + 255) / 256, 256, 0, stream>>>(
      xT, convd_w, convd_b, NUL, NUL, NUL, NUL, c64, 2, 3, 64, 1024);
  conv1d_bn_kernel<<<(2 * 128 * 1024 + 255) / 256, 256, 0, stream>>>(
      c64, conv1d_w, conv1d_b, NUL, NUL, NUL, NUL, c128, 2, 64, 128, 1024);
  conv1d_bn_kernel<<<(2 * 256 * 1024 + 255) / 256, 256, 0, stream>>>(
      c128, c1_w, c1_b, in[7], in[8], in[9], in[10], cb1, 2, 128, 256, 1024); // x1
  conv1d_bn_kernel<<<(2 * 256 * 1024 + 255) / 256, 256, 0, stream>>>(
      cb1, c2_w, c2_b, in[13], in[14], in[15], in[16], cb2, 2, 256, 256, 1024); // x2
  elu_add_kernel<<<(2 * 256 * 1024 + 255) / 256, 256, 0, stream>>>(cb2, cb1, cb2, 2 * 256 * 1024); // out2
  conv1d_bn_kernel<<<(2 * 256 * 1024 + 255) / 256, 256, 0, stream>>>(
      cb2, c3_w, c3_b, in[19], in[20], in[21], in[22], cb1, 2, 256, 256, 1024); // x3
  elu_add_kernel<<<(2 * 256 * 1024 + 255) / 256, 256, 0, stream>>>(cb1, cb2, cb1, 2 * 256 * 1024); // out3
  conv1d_bn_kernel<<<(2 * 256 * 1024 + 255) / 256, 256, 0, stream>>>(
      cb1, c4_w, c4_b, in[25], in[26], in[27], in[28], cb2, 2, 256, 256, 1024); // x4
  elu_add_kernel<<<(2 * 256 * 1024 + 255) / 256, 256, 0, stream>>>(cb2, cb1, cb2, 2 * 256 * 1024); // out4
  to_tokens_kernel<<<(2048 * 256) / 256, 256, 0, stream>>>(cb2, htok);

  // ---- 4 mamba2 layers ----
  for (int li = 0; li < 4; ++li) {
    const float* m_in_w   = in[29 + li * 8 + 0];
    const float* m_conv_w = in[29 + li * 8 + 1];
    const float* m_conv_b = in[29 + li * 8 + 2];
    const float* m_dtb    = in[29 + li * 8 + 3];
    const float* m_Alog   = in[29 + li * 8 + 4];
    const float* m_D      = in[29 + li * 8 + 5];
    const float* m_nw     = in[29 + li * 8 + 6];
    const float* m_out_w  = in[29 + li * 8 + 7];
    const float* bn_g = in[61 + li * 4 + 0];
    const float* bn_b = in[61 + li * 4 + 1];
    const float* bn_m = in[61 + li * 4 + 2];
    const float* bn_v = in[61 + li * 4 + 3];

    gemm(htok, m_in_w, nullptr, zx, 2048, 8512, 256, 0);
    dt_kernel<<<(2048 * 64) / 256, 256, 0, stream>>>(zx, m_dtb, dtp);
    mamba_conv_kernel<<<(int)(((long)2048 * 4352 + 255) / 256), 256, 0, stream>>>(
        zx, m_conv_w, m_conv_b, xbc);
    // G[b,ch] = C_chunk (64x128) * B_chunk^T : batched WMMA, z=(b*16+ch)
    {
      dim3 grid(1, 1, 32);
      gemm_wmma_ex<<<grid, 256, 0, stream>>>(
          xbc + 4224, xbc + 4096, NUL, gbuf,
          64, 64, 128, 4352, 4352, 64,
          4, (long)1024 * 4352, (long)64 * 4352,
             (long)1024 * 4352, (long)64 * 4352,
             (long)16 * 4096,   (long)4096,
          1.0f, 0, 0);
    }
    ssd_kernel<<<128, 256, 0, stream>>>(xbc, dtp, gbuf, m_Alog, m_D, ybig);
    gate_rms_kernel<<<2048, 256, 0, stream>>>(ybig, zx, m_nw);
    gemm(ybig, m_out_w, nullptr, tmp256, 2048, 256, 4096, 0);
    bn_lc_kernel<<<(2048 * 256) / 256, 256, 0, stream>>>(tmp256, bn_g, bn_b, bn_m, bn_v, htok);
  }

  // ---- 2 attention blocks (shared params), WMMA scores + WMMA P*V ----
  const long QKV_B = (long)1024 * 768;      // qkv batch stride (per b)
  for (int rep = 0; rep < 2; ++rep) {
    gemm(htok, attn_in_w, attn_in_b, xbc /*qkv (T,768)*/, 2048, 768, 256, 0);
    // S[z] = Q[z] * K[z]^T / sqrt(32); z = b*8+head
    {
      dim3 grid(1024 / 128, 1024 / 64, 16);
      gemm_wmma_ex<<<grid, 256, 0, stream>>>(
          xbc /*Q*/, xbc + 256 /*K*/, NUL, sbuf,
          1024, 1024, 32, 768, 768, 1024,
          3, QKV_B, 32, QKV_B, 32,
             (long)8 * 1024 * 1024, (long)1024 * 1024,
          0.1767766953f, 0, 0);
    }
    softmax_row_kernel<<<16384, 256, 0, stream>>>(sbuf);
    // O[z] = P[z] * V[z]  (NN: V is K-major), output packed into (T,256)
    {
      dim3 grid(1, 1024 / 64, 16);
      gemm_wmma_ex<<<grid, 256, 0, stream>>>(
          sbuf, xbc + 512 /*V*/, NUL, ybig,
          1024, 32, 1024, 1024, 768, 256,
          3, (long)8 * 1024 * 1024, (long)1024 * 1024,
             QKV_B, 32,
             (long)1024 * 256, 32,
          1.0f, 0, 1);
    }
    gemm(ybig, attn_out_w, attn_out_b, tmp256, 2048, 256, 256, 0);
    add_ln_kernel<<<2048, 256, 0, stream>>>(tmp256, htok, ln_g, ln_b, htok);
    bn_lc_kernel<<<(2048 * 256) / 256, 256, 0, stream>>>(
        htok, in[61], in[62], in[63], in[64], htok); // bn1
  }

  // ---- head ----
  gemm(htok, fc_w, fc_b, fcbuf, 2048, 128, 256, 1 /*relu*/);
  outproj_kernel<<<(2048 * 9 + 255) / 256, 256, 0, stream>>>(fcbuf, ow, ob, (float*)d_out);
  (void)out_size;
}
